// LSTMModel_70093866270906
// MI455X (gfx1250) — matrix operand order repaired and checked by reading, not compiled, not run on hardware
//
#include <hip/hip_runtime.h>

// ---------------------------------------------------------------------------
// Fully-fused 2-layer LSTM for MI455X (gfx1250, wave32, WMMA).
//
// The recurrence is independent per batch row, so each workgroup owns a
// 16-row batch tile and runs ALL 512 timesteps of BOTH layers persistently.
// All four weight matrices live in LDS as f16 WMMA-B fragments (~160KB —
// only possible with CDNA5's 320KB/WGP LDS). The inter-layer activation y0
// and both hidden states never touch global memory; global traffic is the
// L2-resident embedding gathers + indices + 128KB of final state.
// Gates use v_wmma_f32_16x16x32_f16 (f16 in, f32 accumulate); the first WMMA
// of each chain takes a zero C operand (inline-0 SRC2 per ISA) and the bias
// is folded into the activation epilogue.
// ---------------------------------------------------------------------------

typedef __attribute__((ext_vector_type(16))) _Float16 v16h;
typedef __attribute__((ext_vector_type(8)))  _Float16 v8h;
typedef __attribute__((ext_vector_type(8)))  float    v8f;

namespace {
constexpr int Bsz = 512, Sseq = 512, Edim = 128, Hdim = 64, Gdim = 256; // G = 4H
constexpr int Vvoc = 50000;
constexpr int BT   = 16;            // batch rows per workgroup (one WMMA M-tile)
constexpr int WGT  = 128;           // 4 waves (wave32): wave w owns hidden chunk w
constexpr int NWG  = Bsz / BT;      // 32 workgroups

// LDS layout in _Float16 units. Fragment = 512 halves = one 16x32 f16 tile
// stored "fragment-ready": element (lane, j) at [lane*16 + j] so each lane
// reads its v16h operand as 2x ds_load_b128.
constexpr int OFF_WIH0 = 0;                      // 16 N-tiles * 4 K-chunks
constexpr int OFF_WHH0 = OFF_WIH0 + 16 * 4 * 512;
constexpr int OFF_WIH1 = OFF_WHH0 + 16 * 2 * 512;
constexpr int OFF_WHH1 = OFF_WIH1 + 16 * 2 * 512;
constexpr int OFF_XBUF = OFF_WHH1 + 16 * 2 * 512; // x_t A-frags: 4 K-chunks
constexpr int OFF_H0   = OFF_XBUF + 4 * 512;      // h0 A-frags: 2 K-chunks
constexpr int OFF_H1   = OFF_H0   + 2 * 512;      // h1 A-frags: 2 K-chunks
constexpr int HALF_TOT = OFF_H1   + 2 * 512;
constexpr size_t LDS_BYTES = size_t(HALF_TOT) * 2; // 172032 B < 320KB/WGP
}

// A-matrix (16x32 f16) K index for (lane, halfIdx j), per ISA §7.12.2:
//   lanes 0-15 hold K = {0..7,16..23}, lanes 16-31 hold K = {8..15,24..31}
// For j = 8*jh + i (i=0..7):  K = 16*jh + 8*(lane>>4) + i   (8 consecutive K!)
__device__ __forceinline__ int b_k(int lane, int j) {
    // B-matrix (32x16 f16): same K order as A: K = 16*(j>>3) + 8*(lane>>4) + (j&7)
    return (j & 7) + ((lane >> 4) << 3) + ((j >> 3) << 4);
}

__device__ __forceinline__ float sigf(float x) {
    return 1.0f / (1.0f + __expf(-x));
}
__device__ __forceinline__ float tanh_fast(float x) {
    return 2.0f / (1.0f + __expf(-2.0f * x)) - 1.0f; // robust at +-inf
}

// Convert W[Gdim x Kdim] (f32 row-major; gates = A @ W^T so B[k][n] = W[16t+n, k])
// into f16 B-fragment layout in LDS. One-time cost.
__device__ __forceinline__ void stage_weights(const float* __restrict__ W, int Kdim,
                                              _Float16* dst, int tid) {
    const int chunks = Kdim >> 5;
    const int total  = 16 * chunks * 512;
    for (int e = tid; e < total; e += WGT) {
        const int frag = e >> 9;          // = tile*chunks + chunk
        const int r    = e & 511;
        const int ln   = r >> 4, j = r & 15;
        const int t    = frag / chunks, c = frag - t * chunks;
        const int n    = ln & 15;
        const int k    = (c << 5) + b_k(ln, j);
        dst[e] = (_Float16)W[(t * 16 + n) * Kdim + k];
    }
}

__global__ void __launch_bounds__(WGT, 1)
lstm2_fused(const int* __restrict__ iseq, const float* __restrict__ emb,
            const float* __restrict__ Wih0, const float* __restrict__ Whh0,
            const float* __restrict__ bih0, const float* __restrict__ bhh0,
            const float* __restrict__ Wih1, const float* __restrict__ Whh1,
            const float* __restrict__ bih1, const float* __restrict__ bhh1,
            float* __restrict__ h1T)
{
    extern __shared__ _Float16 lds[];

    const int tid  = threadIdx.x;
    const int lane = tid & 31;
    const int wave = tid >> 5;          // hidden chunk owned by this wave (0..3)
    const int b0   = blockIdx.x * BT;   // first batch row of this tile

    // ---- one-time staging: weights -> f16 B-frags, zero h state ----
    stage_weights(Wih0, Edim, lds + OFF_WIH0, tid);
    stage_weights(Whh0, Hdim, lds + OFF_WHH0, tid);
    stage_weights(Wih1, Hdim, lds + OFF_WIH1, tid);
    stage_weights(Whh1, Hdim, lds + OFF_WHH1, tid);
    for (int e = tid; e < 4 * 512; e += WGT) lds[OFF_H0 + e] = (_Float16)0.0f;

    // Per-wave bias (applied in the activation epilogue, not as WMMA C-init):
    // gate-tile for (gate g, chunk w) is T = 4g + w; D-layout lane holds
    // column N = lane&15, so this lane's gate index is 16T + (lane&15).
    float bia0[4], bia1[4];
#pragma unroll
    for (int g = 0; g < 4; ++g) {
        const int gi = 16 * (4 * g + wave) + (lane & 15);
        bia0[g] = bih0[gi] + bhh0[gi];
        bia1[g] = bih1[gi] + bhh1[gi];
    }

    v8f vzero;
#pragma unroll
    for (int v = 0; v < 8; ++v) vzero[v] = 0.0f;
    v8f c0 = vzero, c1 = vzero, h1r = vzero;

    __syncthreads();

    for (int t = 0; t < Sseq; ++t) {
        // ---- stage x_t = f16(emb[idx[b0..b0+15, t]]) into A-frag layout ----
        // 2048 halves = 256 groups of 8 *consecutive* K values: each group is
        // 2x global_load_b128 -> 8 cvt -> 1x ds_store_b128.
        for (int gi = tid; gi < 256; gi += WGT) {
            const int c  = gi >> 6;            // K-chunk (0..3)
            const int r  = gi & 63;
            const int ln = r >> 1;             // lane slot (0..31)
            const int jh = r & 1;              // j half: j0 = 8*jh
            const int m  = ln & 15;            // batch row in tile
            const int k0 = (c << 5) + (jh << 4) + ((ln >> 4) << 3);
            int row = iseq[(b0 + m) * Sseq + t];
            if ((unsigned)row >= (unsigned)Vvoc) row = 0;
            const float4* src = (const float4*)(emb + row * Edim + k0);
            const float4 x0 = src[0], x1 = src[1];
            v8h hv;
            hv[0] = (_Float16)x0.x; hv[1] = (_Float16)x0.y;
            hv[2] = (_Float16)x0.z; hv[3] = (_Float16)x0.w;
            hv[4] = (_Float16)x1.x; hv[5] = (_Float16)x1.y;
            hv[6] = (_Float16)x1.z; hv[7] = (_Float16)x1.w;
            *(v8h*)(lds + OFF_XBUF + (c << 9) + ln * 16 + (jh << 3)) = hv;
        }
        // prefetch next step's embedding rows into cache (global_prefetch_b8)
        if (t + 1 < Sseq && tid < BT) {
            const int nrow = iseq[(b0 + tid) * Sseq + t + 1];
            if ((unsigned)nrow < (unsigned)Vvoc)
                __builtin_prefetch(&emb[nrow * Edim], 0, 3);
        }
        __syncthreads();

        // ================= layer 0: gates = x@Wih0^T + h0@Whh0^T =============
        v8f acc[4];
        {   // first K-chunk: C = zero (inline-0 SRC2 opportunity)
            const v16h xa = *(const v16h*)(lds + OFF_XBUF + lane * 16);
#pragma unroll
            for (int g = 0; g < 4; ++g) {
                const int tile = 4 * g + wave;
                const v16h wb = *(const v16h*)(lds + OFF_WIH0 + (tile * 4) * 512 + lane * 16);
                acc[g] = __builtin_amdgcn_wmma_f32_16x16x32_f16(
                    false, xa, false, wb, (short)0, vzero, false, false);
            }
        }
#pragma unroll
        for (int c = 1; c < 4; ++c) {           // remaining K-chunks of E=128
            const v16h xa = *(const v16h*)(lds + OFF_XBUF + c * 512 + lane * 16);
#pragma unroll
            for (int g = 0; g < 4; ++g) {
                const int tile = 4 * g + wave;
                const v16h wb = *(const v16h*)(lds + OFF_WIH0 + (tile * 4 + c) * 512 + lane * 16);
                acc[g] = __builtin_amdgcn_wmma_f32_16x16x32_f16(
                    false, xa, false, wb, (short)0, acc[g], false, false);
            }
        }
#pragma unroll
        for (int c = 0; c < 2; ++c) {           // recurrent K-chunks of H=64
            const v16h ha = *(const v16h*)(lds + OFF_H0 + c * 512 + lane * 16);
#pragma unroll
            for (int g = 0; g < 4; ++g) {
                const int tile = 4 * g + wave;
                const v16h wb = *(const v16h*)(lds + OFF_WHH0 + (tile * 2 + c) * 512 + lane * 16);
                acc[g] = __builtin_amdgcn_wmma_f32_16x16x32_f16(
                    false, ha, false, wb, (short)0, acc[g], false, false);
            }
        }
        v8f h0r;
#pragma unroll
        for (int v = 0; v < 8; ++v) {
            const float iv = sigf(acc[0][v] + bia0[0]);
            const float fv = sigf(acc[1][v] + bia0[1]);
            const float gv = tanh_fast(acc[2][v] + bia0[2]);
            const float ov = sigf(acc[3][v] + bia0[3]);
            const float cv = fv * c0[v] + iv * gv;
            c0[v] = cv;
            h0r[v] = ov * tanh_fast(cv);
        }
        __syncthreads();                         // everyone done reading old h0
        // D-layout -> A-frag layout (f16) for h0 (feeds layer 1 and next step)
#pragma unroll
        for (int v = 0; v < 8; ++v) {
            const int m  = v + ((lane >> 4) << 3);   // batch row
            const int hh = 16 * wave + (lane & 15);  // hidden index 0..63
            const int cc = hh >> 5, kk = hh & 31;
            const int g2 = (kk >> 3) & 1;
            const int j  = (kk & 7) + ((kk >> 4) << 3);
            lds[OFF_H0 + cc * 512 + (m + 16 * g2) * 16 + j] = (_Float16)h0r[v];
        }
        __syncthreads();

        // ================= layer 1: gates = h0@Wih1^T + h1@Whh1^T ============
        {   // first K-chunk: C = zero
            const v16h xa = *(const v16h*)(lds + OFF_H0 + lane * 16);
#pragma unroll
            for (int g = 0; g < 4; ++g) {
                const int tile = 4 * g + wave;
                const v16h wb = *(const v16h*)(lds + OFF_WIH1 + (tile * 2) * 512 + lane * 16);
                acc[g] = __builtin_amdgcn_wmma_f32_16x16x32_f16(
                    false, xa, false, wb, (short)0, vzero, false, false);
            }
        }
        {
            const v16h xa = *(const v16h*)(lds + OFF_H0 + 512 + lane * 16);
#pragma unroll
            for (int g = 0; g < 4; ++g) {
                const int tile = 4 * g + wave;
                const v16h wb = *(const v16h*)(lds + OFF_WIH1 + (tile * 2 + 1) * 512 + lane * 16);
                acc[g] = __builtin_amdgcn_wmma_f32_16x16x32_f16(
                    false, xa, false, wb, (short)0, acc[g], false, false);
            }
        }
#pragma unroll
        for (int c = 0; c < 2; ++c) {
            const v16h ha = *(const v16h*)(lds + OFF_H1 + c * 512 + lane * 16);
#pragma unroll
            for (int g = 0; g < 4; ++g) {
                const int tile = 4 * g + wave;
                const v16h wb = *(const v16h*)(lds + OFF_WHH1 + (tile * 2 + c) * 512 + lane * 16);
                acc[g] = __builtin_amdgcn_wmma_f32_16x16x32_f16(
                    false, ha, false, wb, (short)0, acc[g], false, false);
            }
        }
#pragma unroll
        for (int v = 0; v < 8; ++v) {
            const float iv = sigf(acc[0][v] + bia1[0]);
            const float fv = sigf(acc[1][v] + bia1[1]);
            const float gv = tanh_fast(acc[2][v] + bia1[2]);
            const float ov = sigf(acc[3][v] + bia1[3]);
            const float cv = fv * c1[v] + iv * gv;
            c1[v] = cv;
            h1r[v] = ov * tanh_fast(cv);
        }
        __syncthreads();                         // everyone done reading old h1
#pragma unroll
        for (int v = 0; v < 8; ++v) {
            const int m  = v + ((lane >> 4) << 3);
            const int hh = 16 * wave + (lane & 15);
            const int cc = hh >> 5, kk = hh & 31;
            const int g2 = (kk >> 3) & 1;
            const int j  = (kk & 7) + ((kk >> 4) << 3);
            lds[OFF_H1 + cc * 512 + (m + 16 * g2) * 16 + j] = (_Float16)h1r[v];
        }
        __syncthreads();
    }

    // ---- final hidden state (full f32 precision from registers) -> global ----
#pragma unroll
    for (int v = 0; v < 8; ++v) {
        const int m  = v + ((lane >> 4) << 3);
        const int hh = 16 * wave + (lane & 15);
        h1T[(b0 + m) * Hdim + hh] = h1r[v];
    }
}

// out = sigmoid(h1T @ Wfc^T + bfc), [512, 2] — trivially small, pure VALU.
__global__ void fc_out(const float* __restrict__ h1T, const float* __restrict__ Wfc,
                       const float* __restrict__ bfc, float* __restrict__ out)
{
    const int b = blockIdx.x * blockDim.x + threadIdx.x;
    if (b >= Bsz) return;
#pragma unroll
    for (int o = 0; o < 2; ++o) {
        float s = bfc[o];
#pragma unroll 8
        for (int k = 0; k < Hdim; ++k) s += h1T[b * Hdim + k] * Wfc[o * Hdim + k];
        out[b * 2 + o] = sigf(s);
    }
}

extern "C" void kernel_launch(void* const* d_in, const int* in_sizes, int n_in,
                              void* d_out, int out_size, void* d_ws, size_t ws_size,
                              hipStream_t stream)
{
    (void)in_sizes; (void)n_in; (void)out_size; (void)ws_size;
    const int*   iseq = (const int*)d_in[0];
    const float* emb  = (const float*)d_in[1];
    const float* Wih0 = (const float*)d_in[2];
    const float* Whh0 = (const float*)d_in[3];
    const float* bih0 = (const float*)d_in[4];
    const float* bhh0 = (const float*)d_in[5];
    const float* Wih1 = (const float*)d_in[6];
    const float* Whh1 = (const float*)d_in[7];
    const float* bih1 = (const float*)d_in[8];
    const float* bhh1 = (const float*)d_in[9];
    const float* Wfc  = (const float*)d_in[10];
    const float* bfc  = (const float*)d_in[11];
    float* out = (float*)d_out;
    float* h1T = (float*)d_ws;          // [512 x 64] f32 scratch (128 KB)

    // Allow >48KB dynamic LDS (CDNA5 WGP has 320KB). Host-side attribute set,
    // deterministic, not a stream op (graph-capture safe).
    (void)hipFuncSetAttribute((const void*)lstm2_fused,
                              hipFuncAttributeMaxDynamicSharedMemorySize,
                              (int)LDS_BYTES);

    lstm2_fused<<<dim3(NWG), dim3(WGT), LDS_BYTES, stream>>>(
        iseq, emb, Wih0, Whh0, bih0, bhh0, Wih1, Whh1, bih1, bhh1, h1T);

    fc_out<<<dim3((Bsz + 255) / 256), dim3(256), 0, stream>>>(h1T, Wfc, bfc, out);
}
